// DiaM_55490977464516
// MI455X (gfx1250) — matrix operand order repaired and checked by reading, not compiled, not run on hardware
//
#include <hip/hip_runtime.h>
#include <hip/hip_bf16.h>

typedef __bf16 bf16;
typedef __attribute__((ext_vector_type(16))) __bf16 v16bf;
typedef __attribute__((ext_vector_type(8)))  __bf16 v8bf;
typedef __attribute__((ext_vector_type(8)))  float   v8f;
typedef __attribute__((ext_vector_type(4)))  float   v4f;

#define B_ 256
#define C_ 384
#define D_ 384

union ABfrag { v16bf v; v8bf h[2]; };

// load 8 consecutive f32 and convert to 8 bf16
__device__ __forceinline__ v8bf cvt8(const float* __restrict__ p) {
    v4f a = *(const v4f*)p;
    v4f b = *(const v4f*)(p + 4);
    v8bf r;
#pragma unroll
    for (int i = 0; i < 4; ++i) { r[i] = (bf16)a[i]; r[i + 4] = (bf16)b[i]; }
    return r;
}

// ---------------------------------------------------------------------------
// QKV projection: y = z @ W^T + b  (bf16 output).
// transposed==0 : dst is row-major bf16 [B*C, D]   (Q, K)
// transposed==1 : dst is bf16 [B, D, C]            (V^T, for PV B-fragments)
// Block: 256 thr = 8 waves; block tile 128 rows x 128 cols; wave tile 16x128.
// ---------------------------------------------------------------------------
__global__ __launch_bounds__(256)
void proj_kernel(const float* __restrict__ z, const float* __restrict__ W,
                 const float* __restrict__ bias, bf16* __restrict__ dst,
                 int transposed)
{
    const int lane = threadIdx.x & 31;
    const int wave = threadIdx.x >> 5;
    const int half = lane >> 4;
    const int ln   = lane & 15;
    const int n0   = blockIdx.x * 128;
    const int r0   = blockIdx.y * 128 + wave * 16;

    v8f acc[8];
#pragma unroll
    for (int i = 0; i < 8; ++i) acc[i] = 0.0f;

    const float* zrow = z + (size_t)(r0 + ln) * D_;

    for (int k0 = 0; k0 < D_; k0 += 32) {
        // A-frag (16x32 bf16): lane ln holds row ln; e0..7 -> k0+8*half.., e8..15 -> +16
        ABfrag a;
        a.h[0] = cvt8(zrow + k0 + 8 * half);
        a.h[1] = cvt8(zrow + k0 + 8 * half + 16);
#pragma unroll
        for (int nt = 0; nt < 8; ++nt) {
            // B-frag (32x16): B[k][n] = W[n][k]; lane col n, e0..7 -> k0+8*half.., e8..15 -> +16
            const float* wrow = W + (size_t)(n0 + nt * 16 + ln) * D_ + k0 + 8 * half;
            ABfrag bfr;
            bfr.h[0] = cvt8(wrow);
            bfr.h[1] = cvt8(wrow + 16);
            acc[nt] = __builtin_amdgcn_wmma_f32_16x16x32_bf16(
                false, a.v, false, bfr.v, (short)0, acc[nt], false, false);
        }
    }

    if (!transposed) {
#pragma unroll
        for (int nt = 0; nt < 8; ++nt) {
            const int n = n0 + nt * 16 + ln;
            const float bv = bias[n];
#pragma unroll
            for (int r = 0; r < 8; ++r) {
                const int row = r0 + r + 8 * half;        // C layout: row r+8*half, col n
                dst[(size_t)row * D_ + n] = (bf16)(acc[nt][r] + bv);
            }
        }
    } else {
        const int bidx = r0 / C_;                          // 128 | 384 -> no batch straddle
        const int cc0  = (r0 - bidx * C_) + 8 * half;      // 8 consecutive tokens per lane
#pragma unroll
        for (int nt = 0; nt < 8; ++nt) {
            const int e = n0 + nt * 16 + ln;
            const float bv = bias[e];
            v8bf pack;
#pragma unroll
            for (int r = 0; r < 8; ++r) pack[r] = (bf16)(acc[nt][r] + bv);
            *(v8bf*)(dst + ((size_t)bidx * D_ + e) * C_ + cc0) = pack;
        }
    }
}

// ---------------------------------------------------------------------------
// Causal flash attention. Grid (C/128 q-blocks, B). 8 waves; wave = 16 q rows.
// Q,K row-major bf16 [B*C, D]; Vt bf16 [B, D, C]; out f32 [B*C, D].
// K/V chunks staged in LDS once per block (8x reduction in global operand
// traffic); B-fragments fed from ds_load_b128.
// ---------------------------------------------------------------------------
__global__ __launch_bounds__(256)
void attn_kernel(const bf16* __restrict__ Q, const bf16* __restrict__ K,
                 const bf16* __restrict__ Vt, float* __restrict__ out)
{
    __shared__ bf16 kTile[32 * 384];        // [tok][d]   24KB, contiguous copy
    __shared__ bf16 vTile[384 * 32];        // [d][kv]    24KB
    __shared__ bf16 pScr[8 * 16 * 32];      // per-wave P transpose scratch, 8KB

    const int lane = threadIdx.x & 31;
    const int wave = threadIdx.x >> 5;
    const int half = lane >> 4;
    const int ln   = lane & 15;
    const int b    = blockIdx.y;
    const int q0   = blockIdx.x * 128;
    const int qw   = q0 + wave * 16;        // this wave's first query row
    const int kvEnd = q0 + 128;             // block-uniform causal bound (exclusive)

    // Preload all Q A-fragments for the wave's 16 rows (12 x K=32)
    ABfrag aq[12];
    {
        const bf16* qrow = Q + ((size_t)b * C_ + qw + ln) * D_;
#pragma unroll
        for (int kk = 0; kk < 12; ++kk) {
            aq[kk].h[0] = *(const v8bf*)(qrow + kk * 32 + 8 * half);
            aq[kk].h[1] = *(const v8bf*)(qrow + kk * 32 + 8 * half + 16);
        }
    }

    v8f o[24];
#pragma unroll
    for (int i = 0; i < 24; ++i) o[i] = 0.0f;
    float mrow[8], lrow[8];
#pragma unroll
    for (int r = 0; r < 8; ++r) { mrow[r] = -3.0e38f; lrow[r] = 0.0f; }

    const float inv_sqrt_d = 0.05103103630798288f;        // 1/sqrt(384)
    bf16* myP = pScr + wave * (16 * 32);

    for (int kv0 = 0; kv0 < kvEnd; kv0 += 32) {
        // ---- cooperative LDS staging (all 256 threads) ----
        {
            // K chunk: 32 consecutive token rows -> one linear 24KB copy
            const uint4* ksrc = (const uint4*)(K + ((size_t)b * C_ + kv0) * D_);
            uint4* kdst = (uint4*)kTile;
            for (int i = threadIdx.x; i < (32 * 384) / 8; i += 256)
                kdst[i] = ksrc[i];
            // Vt chunk: 384 rows x 64B, row stride C_ in global
            const uint4* vsrc = (const uint4*)(Vt + (size_t)b * D_ * C_);
            uint4* vdst = (uint4*)vTile;
            for (int i = threadIdx.x; i < (384 * 32) / 8; i += 256) {
                const int row = i >> 2, q4 = i & 3;       // 4 uint4 per 32-elem row
                vdst[i] = vsrc[row * (C_ / 8) + (kv0 >> 3) + q4];
            }
        }
        __syncthreads();

        if (kv0 <= qw + 15) {                             // per-wave causal pruning
            // ---- S = Q @ K^T for this 16x32 chunk (B-frags from LDS) ----
            v8f s[2];
            s[0] = 0.0f; s[1] = 0.0f;
#pragma unroll
            for (int kk = 0; kk < 12; ++kk) {
#pragma unroll
                for (int t = 0; t < 2; ++t) {
                    const bf16* krow = kTile + (t * 16 + ln) * D_ + kk * 32 + 8 * half;
                    ABfrag bk;
                    bk.h[0] = *(const v8bf*)(krow);
                    bk.h[1] = *(const v8bf*)(krow + 16);
                    s[t] = __builtin_amdgcn_wmma_f32_16x16x32_bf16(
                        false, aq[kk].v, false, bk.v, (short)0, s[t], false, false);
                }
            }
            // ---- causal mask + scale + online softmax ----
            float p0[8], p1[8], rmax[8];
#pragma unroll
            for (int r = 0; r < 8; ++r) {
                const int q = qw + r + 8 * half;
                const float v0 = (kv0 + ln      <= q) ? s[0][r] * inv_sqrt_d : -1.0e30f;
                const float v1 = (kv0 + 16 + ln <= q) ? s[1][r] * inv_sqrt_d : -1.0e30f;
                p0[r] = v0; p1[r] = v1;
                rmax[r] = fmaxf(v0, v1);
            }
#pragma unroll
            for (int off = 1; off < 16; off <<= 1)        // stays inside half-wave
#pragma unroll
                for (int r = 0; r < 8; ++r)
                    rmax[r] = fmaxf(rmax[r], __shfl_xor(rmax[r], off, 32));

            float alpha[8], rsum[8];
#pragma unroll
            for (int r = 0; r < 8; ++r) {
                const float mnew = fmaxf(mrow[r], rmax[r]);
                alpha[r] = __expf(mrow[r] - mnew);
                mrow[r]  = mnew;
                p0[r] = __expf(p0[r] - mnew);
                p1[r] = __expf(p1[r] - mnew);
                rsum[r] = p0[r] + p1[r];
            }
#pragma unroll
            for (int off = 1; off < 16; off <<= 1)
#pragma unroll
                for (int r = 0; r < 8; ++r)
                    rsum[r] += __shfl_xor(rsum[r], off, 32);
#pragma unroll
            for (int r = 0; r < 8; ++r) lrow[r] = lrow[r] * alpha[r] + rsum[r];
#pragma unroll
            for (int i = 0; i < 24; ++i)
#pragma unroll
                for (int r = 0; r < 8; ++r) o[i][r] *= alpha[r];

            // ---- transpose P (C layout -> A layout) through LDS ----
#pragma unroll
            for (int r = 0; r < 8; ++r) {
                myP[(r + 8 * half) * 32 + ln]      = (bf16)p0[r];
                myP[(r + 8 * half) * 32 + 16 + ln] = (bf16)p1[r];
            }
            asm volatile("s_wait_dscnt 0" ::: "memory");
            ABfrag pa;
            pa.h[0] = *(const v8bf*)(myP + ln * 32 + 8 * half);
            pa.h[1] = *(const v8bf*)(myP + ln * 32 + 8 * half + 16);

            // ---- O += P @ V_chunk (B-frags from LDS vTile[d][kv]) ----
#pragma unroll
            for (int nt = 0; nt < 24; ++nt) {
                const bf16* vrow = vTile + (nt * 16 + ln) * 32 + 8 * half;
                ABfrag bv;
                bv.h[0] = *(const v8bf*)(vrow);
                bv.h[1] = *(const v8bf*)(vrow + 16);
                o[nt] = __builtin_amdgcn_wmma_f32_16x16x32_bf16(
                    false, pa.v, false, bv.v, (short)0, o[nt], false, false);
            }
        }
        __syncthreads();                                  // protect LDS before next copy
    }

    // ---- finalize: divide by softmax denominator, store f32 ----
    float rcp[8];
#pragma unroll
    for (int r = 0; r < 8; ++r) rcp[r] = 1.0f / lrow[r];
#pragma unroll
    for (int nt = 0; nt < 24; ++nt) {
        const int n = nt * 16 + ln;
#pragma unroll
        for (int r = 0; r < 8; ++r) {
            const int q = qw + r + 8 * half;
            out[((size_t)b * C_ + q) * D_ + n] = o[nt][r] * rcp[r];
        }
    }
}

extern "C" void kernel_launch(void* const* d_in, const int* in_sizes, int n_in,
                              void* d_out, int out_size, void* d_ws, size_t ws_size,
                              hipStream_t stream) {
    const float* z  = (const float*)d_in[0];
    const float* Wq = (const float*)d_in[1];
    const float* bq = (const float*)d_in[2];
    const float* Wk = (const float*)d_in[3];
    const float* bk = (const float*)d_in[4];
    const float* Wv = (const float*)d_in[5];
    const float* bv = (const float*)d_in[6];
    float* out = (float*)d_out;

    const size_t nElems = (size_t)B_ * C_ * D_;
    bf16* Qbf = (bf16*)d_ws;
    bf16* Kbf = Qbf + nElems;
    bf16* Vt  = Kbf + nElems;

    dim3 pgrid(D_ / 128, (B_ * C_) / 128);
    proj_kernel<<<pgrid, 256, 0, stream>>>(z, Wq, bq, Qbf, 0);
    proj_kernel<<<pgrid, 256, 0, stream>>>(z, Wk, bk, Kbf, 0);
    proj_kernel<<<pgrid, 256, 0, stream>>>(z, Wv, bv, Vt, 1);

    dim3 agrid(C_ / 128, B_);
    attn_kernel<<<agrid, 256, 0, stream>>>(Qbf, Kbf, Vt, out);
}
